// Net_65446711657118
// MI455X (gfx1250) — matrix operand order repaired and checked by reading, not compiled, not run on hardware
//
#include <hip/hip_runtime.h>

// ---------------------------------------------------------------------------
// 2-layer GCN for MI455X (gfx1250, wave32).
//   layer1: agg(norm, x @ W1) + b1 -> relu
//   layer2: agg(norm, h @ W2) + b2 -> log_softmax
// GEMMs use v_wmma_f32_16x16x32_bf16. Edge scatter is atomic-f32 bound
// (~8-9 GB of traffic @ 23.3 TB/s dominates; GEMM FLOPs are trivial on WMMA).
// ---------------------------------------------------------------------------

typedef __attribute__((ext_vector_type(16))) __bf16 bf16x16;
typedef __attribute__((ext_vector_type(8)))  float  f32x8;

union FragBF { bf16x16 v; uint4 q[2]; };

__device__ __forceinline__ unsigned short f2bf(float f) {
    unsigned int u = __float_as_uint(f);
    unsigned int r = u + 0x7FFFu + ((u >> 16) & 1u);   // round-to-nearest-even
    return (unsigned short)(r >> 16);
}

// ---- degree / normalization -----------------------------------------------
__global__ void k_init_cnt(int* cnt, int n) {
    int i = blockIdx.x * blockDim.x + threadIdx.x;
    if (i < n) cnt[i] = 1;                 // self-loop contributes 1
}

__global__ void k_count(const int* __restrict__ dst, int* __restrict__ cnt, int e) {
    int i = blockIdx.x * blockDim.x + threadIdx.x;
    if (i < e) atomicAdd(&cnt[dst[i]], 1);
}

__global__ void k_dis(const int* __restrict__ cnt, float* __restrict__ dis, int n) {
    int i = blockIdx.x * blockDim.x + threadIdx.x;
    if (i < n) dis[i] = rsqrtf((float)cnt[i]);
}

// ---- conversions ----------------------------------------------------------
__global__ void k_cvt_f32_bf16_x4(const float* __restrict__ in,
                                  unsigned short* __restrict__ out, long long total4) {
    long long i = (long long)blockIdx.x * blockDim.x + threadIdx.x;
    if (i >= total4) return;
    const float4 v = *(const float4*)(in + i * 4);
    ushort4 o;
    o.x = f2bf(v.x); o.y = f2bf(v.y); o.z = f2bf(v.z); o.w = f2bf(v.w);
    *(ushort4*)(out + i * 4) = o;
}

// W [K x N] f32 row-major -> Wt [N x K] bf16 row-major
__global__ void k_transpose_w(const float* __restrict__ w,
                              unsigned short* __restrict__ wt, int K, int N) {
    int i = blockIdx.x * blockDim.x + threadIdx.x;
    if (i >= K * N) return;
    int k = i / N, n = i - k * N;
    wt[(size_t)n * K + k] = f2bf(w[i]);
}

__global__ void k_zero(float* p, long long total) {
    long long i = (long long)blockIdx.x * blockDim.x + threadIdx.x;
    if (i < total) p[i] = 0.0f;
}

// ---- WMMA GEMM: C[M,N] = A[M,K](bf16) * Bt[N,K](bf16)^T -------------------
// Block = 256 threads = 8 waves; each wave owns one 16x16 tile of a 32x64
// block tile. Fragment loads follow ISA 7.12.2 layouts (K-contiguous chunks).
__global__ void k_gemm_bf16(const unsigned short* __restrict__ A,
                            const unsigned short* __restrict__ Bt,
                            float* __restrict__ C, int M, int N, int K) {
    const int wave = threadIdx.x >> 5;
    const int lane = threadIdx.x & 31;
    const int wm = wave >> 2;              // 0..1
    const int wn = wave & 3;               // 0..3
    const int mo = blockIdx.y * 32 + wm * 16;
    const int no = blockIdx.x * 64 + wn * 16;
    if (mo + 16 > M || no + 16 > N) return;   // wave-uniform guard (EXEC all 1s)

    const int ml = lane & 15;              // row (A) / col (B,C) within tile
    const int kh = lane >> 4;              // K-half select

    f32x8 acc = {};
    const unsigned short* arow = A  + (size_t)(mo + ml) * K;
    const unsigned short* brow = Bt + (size_t)(no + ml) * K;

    for (int ko = 0; ko < K; ko += 32) {
        FragBF a, b;
        // A 16x32 bf16: lane group kh=0 -> K {0..7,16..23}; kh=1 -> K {8..15,24..31}
        a.q[0] = *(const uint4*)(arow + ko + kh * 8);
        a.q[1] = *(const uint4*)(arow + ko + 16 + kh * 8);
        // B 32x16 bf16: lane group kh=0 -> K {0..7,16..23}; kh=1 -> K {8..15,24..31} (col = no+ml)
        b.q[0] = *(const uint4*)(brow + ko + kh * 8);
        b.q[1] = *(const uint4*)(brow + ko + 16 + kh * 8);
        acc = __builtin_amdgcn_wmma_f32_16x16x32_bf16(
            /*neg_a=*/false, a.v, /*neg_b=*/false, b.v,
            /*c_mod=*/(short)0, acc, /*reuse_a=*/false, /*reuse_b=*/false);
    }

    // C/D 16x16 f32: lane n = ml, VGPR r -> M = (lane>>4)*8 + r
    float* crow = C + (size_t)(mo + kh * 8) * N + no + ml;
#pragma unroll
    for (int r = 0; r < 8; ++r) crow[(size_t)r * N] = acc[r];
}

// ---- edge scatter: agg[dst] += h[src] * dis[src]*dis[dst] -----------------
// one thread per (edge, 4-feature chunk); gshift = log2(F/4)
__global__ void k_scatter(const float* __restrict__ h, const int* __restrict__ src,
                          const int* __restrict__ dst, const float* __restrict__ dis,
                          float* __restrict__ agg, int E, int F, int gshift) {
    long long idx = (long long)blockIdx.x * blockDim.x + threadIdx.x;
    int e = (int)(idx >> gshift);
    if (e >= E) return;
    int g = (int)(idx & ((1 << gshift) - 1));
    int s = src[e], d = dst[e];
    float wgt = dis[s] * dis[d];
    const float4 hv = *(const float4*)(h + (size_t)s * F + (g << 2));
    float* ap = agg + (size_t)d * F + (g << 2);
    atomicAdd(ap + 0, hv.x * wgt);
    atomicAdd(ap + 1, hv.y * wgt);
    atomicAdd(ap + 2, hv.z * wgt);
    atomicAdd(ap + 3, hv.w * wgt);
}

// ---- layer-1 epilogue: + self-loop + bias, relu, requantize to bf16 -------
__global__ void k_bias_relu_cvt(const float* __restrict__ agg, const float* __restrict__ h,
                                const float* __restrict__ dis, const float* __restrict__ b,
                                unsigned short* __restrict__ outb, long long total) {
    long long i = (long long)blockIdx.x * blockDim.x + threadIdx.x;
    if (i >= total) return;
    int n = (int)(i >> 8);                 // F = 256
    int f = (int)(i & 255);
    float dd = dis[n];
    float v = agg[i] + h[i] * dd * dd + b[f];
    outb[i] = f2bf(fmaxf(v, 0.0f));
}

// ---- layer-2 epilogue: + self-loop + bias, log_softmax (64 classes) -------
// one wave32 per node, 2 features per lane
__global__ void k_final(const float* __restrict__ h2, const float* __restrict__ agg2,
                        const float* __restrict__ dis, const float* __restrict__ b2,
                        float* __restrict__ out, int n_nodes) {
    int node = (int)((blockIdx.x * (long long)blockDim.x + threadIdx.x) >> 5);
    int lane = threadIdx.x & 31;
    if (node >= n_nodes) return;
    float dd = dis[node]; dd *= dd;
    size_t base = (size_t)node * 64;
    float v0 = agg2[base + lane]      + h2[base + lane]      * dd + b2[lane];
    float v1 = agg2[base + lane + 32] + h2[base + lane + 32] * dd + b2[lane + 32];
    float m = fmaxf(v0, v1);
#pragma unroll
    for (int o = 16; o > 0; o >>= 1) m = fmaxf(m, __shfl_xor(m, o, 32));
    float s = __expf(v0 - m) + __expf(v1 - m);
#pragma unroll
    for (int o = 16; o > 0; o >>= 1) s += __shfl_xor(s, o, 32);
    float ls = __logf(s);
    out[base + lane]      = v0 - m - ls;
    out[base + lane + 32] = v1 - m - ls;
}

// ---------------------------------------------------------------------------
extern "C" void kernel_launch(void* const* d_in, const int* in_sizes, int n_in,
                              void* d_out, int out_size, void* d_ws, size_t ws_size,
                              hipStream_t stream) {
    const float* x  = (const float*)d_in[0];
    const int*   ei = (const int*)d_in[1];
    const float* W1 = (const float*)d_in[2];
    const float* b1 = (const float*)d_in[3];
    const float* W2 = (const float*)d_in[4];
    const float* b2 = (const float*)d_in[5];
    float* out = (float*)d_out;

    const int D0 = 512, D1 = 256, D2 = 64;
    const int N = in_sizes[0] / D0;        // 100000
    const int E = in_sizes[1] / 2;         // 3200000
    const int* src = ei;
    const int* dst = ei + E;

    auto al = [](size_t v) { return (v + 255) & ~(size_t)255; };
    char* w = (char*)d_ws;
    size_t off = 0;
    int*            cnt  = (int*)(w + off);            off += al((size_t)N * 4);
    float*          dis  = (float*)(w + off);          off += al((size_t)N * 4);
    unsigned short* w1t  = (unsigned short*)(w + off); off += al((size_t)D1 * D0 * 2);
    unsigned short* w2t  = (unsigned short*)(w + off); off += al((size_t)D2 * D1 * 2);
    unsigned short* xb   = (unsigned short*)(w + off); off += al((size_t)N * D0 * 2);
    float*          h1   = (float*)(w + off);          off += al((size_t)N * D1 * 4);
    float*          agg1 = (float*)(w + off);          off += al((size_t)N * D1 * 4);
    // region reuse after layer-1 epilogue:
    unsigned short* a1b  = xb;                               // N*D1 bf16 fits in xb
    float*          h2   = agg1;                             // N*D2 f32
    float*          agg2 = agg1 + (size_t)N * D2;            // N*D2 f32

    auto cdiv = [](long long a, long long b) { return (unsigned int)((a + b - 1) / b); };
    const int T = 256;

    // normalization
    k_init_cnt<<<cdiv(N, T), T, 0, stream>>>(cnt, N);
    k_count   <<<cdiv(E, T), T, 0, stream>>>(dst, cnt, E);
    k_dis     <<<cdiv(N, T), T, 0, stream>>>(cnt, dis, N);

    // conversions
    k_cvt_f32_bf16_x4<<<cdiv((long long)N * D0 / 4, T), T, 0, stream>>>(x, xb, (long long)N * D0 / 4);
    k_transpose_w<<<cdiv(D0 * D1, T), T, 0, stream>>>(W1, w1t, D0, D1);
    k_transpose_w<<<cdiv(D1 * D2, T), T, 0, stream>>>(W2, w2t, D1, D2);

    // layer 1: h1 = x @ W1 (WMMA), scatter, epilogue -> a1b (bf16)
    {
        dim3 grid(D1 / 64, cdiv(N, 32));
        k_gemm_bf16<<<grid, T, 0, stream>>>(xb, w1t, h1, N, D1, D0);
    }
    k_zero<<<cdiv((long long)N * D1, T), T, 0, stream>>>(agg1, (long long)N * D1);
    k_scatter<<<cdiv((long long)E * (D1 / 4), T), T, 0, stream>>>(h1, src, dst, dis, agg1, E, D1, 6);
    k_bias_relu_cvt<<<cdiv((long long)N * D1, T), T, 0, stream>>>(agg1, h1, dis, b1, a1b, (long long)N * D1);

    // layer 2: h2 = a1 @ W2 (WMMA), scatter, log_softmax -> out
    {
        dim3 grid(D2 / 64, cdiv(N, 32));
        k_gemm_bf16<<<grid, T, 0, stream>>>(a1b, w2t, h2, N, D2, D1);
    }
    k_zero<<<cdiv((long long)N * D2, T), T, 0, stream>>>(agg2, (long long)N * D2);
    k_scatter<<<cdiv((long long)E * (D2 / 4), T), T, 0, stream>>>(h2, src, dst, dis, agg2, E, D2, 4);
    k_final<<<cdiv((long long)N * 32, T), T, 0, stream>>>(h2, agg2, dis, b2, out, N);
}
